// LSTMHead_21534966022642
// MI455X (gfx1250) — matrix operand order repaired and checked by reading, not compiled, not run on hardware
//
#include <hip/hip_runtime.h>
#include <hip/hip_bf16.h>

typedef __attribute__((ext_vector_type(16))) _Float16 v16h;
typedef __attribute__((ext_vector_type(8)))  _Float16 v8h;
typedef __attribute__((ext_vector_type(8)))  float    v8f;

union F8  { v8f  v; float     f[8];  };
union H16 { v16h v; _Float16  e[16]; v8h h8[2]; };

__device__ __forceinline__ v8f wmma_f16(v16h a, v16h b, v8f c) {
  // D = A(16x32 f16) * B(32x16 f16) + C(f32)
  return __builtin_amdgcn_wmma_f32_16x16x32_f16(
      /*neg_a=*/false, a, /*neg_b=*/false, b,
      /*c_mod=*/(short)0, c, /*reuse_a=*/false, /*reuse_b=*/false);
}

__device__ __forceinline__ float sigm(float x) { return 1.0f / (1.0f + __expf(-x)); }

// One async 16B global->LDS transfer; IOFFSET applies to BOTH the LDS and the
// global address (ISA 15.18.3 pseudocode), so one LDS-addr VGPR + one vaddr
// VGPR covers the whole slice with immediate offsets. Tracked by ASYNCcnt.
#define ASYNC_G2L_B128(ldsa, vofs, sbase, IMM)                                 \
  asm volatile("global_load_async_to_lds_b128 %0, %1, %2 offset:" #IMM         \
               :: "v"(ldsa), "v"(vofs), "s"(sbase) : "memory")

// Stage one 16KB gate slice (16 rows x 256 f32) with 8 B128 ops per lane.
__device__ __forceinline__ void stage_gates(unsigned ldsa, unsigned vofs,
                                            const float* sbase) {
  ASYNC_G2L_B128(ldsa, vofs, sbase, 0);
  ASYNC_G2L_B128(ldsa, vofs, sbase, 2048);
  ASYNC_G2L_B128(ldsa, vofs, sbase, 4096);
  ASYNC_G2L_B128(ldsa, vofs, sbase, 6144);
  ASYNC_G2L_B128(ldsa, vofs, sbase, 8192);
  ASYNC_G2L_B128(ldsa, vofs, sbase, 10240);
  ASYNC_G2L_B128(ldsa, vofs, sbase, 12288);
  ASYNC_G2L_B128(ldsa, vofs, sbase, 14336);
}

// ---------------- fp32 -> f16 weight conversion ----------------
__global__ void cvt_f16_kernel(const float* __restrict__ src, _Float16* __restrict__ dst, int n) {
  int i = blockIdx.x * blockDim.x + threadIdx.x;
  if (i < n) dst[i] = (_Float16)src[i];
}

// ---------------- layer-0 input gates: G = avgpool2(X) @ Wih0^T + b0 ----------------
// G layout: [T=4096][B=64][4H=256] fp32.  One block per t, 8 waves.
__global__ __launch_bounds__(256) void gates0_kernel(
    const float* __restrict__ XA, const float* __restrict__ XB,
    const _Float16* __restrict__ Wih, const float* __restrict__ bias,
    float* __restrict__ G)
{
  const int t = blockIdx.x;
  const int tid = threadIdx.x, lane = tid & 31, w = tid >> 5;
  const int m = w >> 1, nh = w & 1;       // m: batch tile (0..3), nh: column half
  const int ln = lane & 15, hh = lane >> 4;
  const int b = 16 * m + ln;              // A-matrix row = batch index

  // source rows 2t, 2t+1 of concat(XA,XB); both fall in the same segment
  const int s0 = 2 * t;
  const float* r0;
  if (s0 < 2048) r0 = XA + ((size_t)b * 2048 + s0) * 128;
  else           r0 = XB + ((size_t)b * 6144 + (s0 - 2048)) * 128;
  const float* r1 = r0 + 128;

  F8 acc[8];
#pragma unroll
  for (int j = 0; j < 8; j++)
#pragma unroll
    for (int v = 0; v < 8; v++) acc[j].f[v] = 0.0f;

#pragma unroll
  for (int s = 0; s < 4; s++) {           // K = 128 = 4 x 32
    const int k0 = 32 * s;
    H16 a;
#pragma unroll
    for (int j = 0; j < 8; j++) {         // fused downsample + f32->f16
      a.e[j]     = (_Float16)(0.5f * (r0[k0 + hh * 16 + j]      + r1[k0 + hh * 16 + j]));
      a.e[8 + j] = (_Float16)(0.5f * (r0[k0 + hh * 16 + 8 + j] + r1[k0 + hh * 16 + 8 + j]));
    }
#pragma unroll
    for (int j = 0; j < 8; j++) {
      const int col0 = (nh * 8 + j) * 16;
      v16h bf = *(const v16h*)(Wih + (size_t)(col0 + ln) * 128 + k0 + hh * 16);
      acc[j].v = wmma_f16(a.v, bf, acc[j].v);
    }
  }

  float* gp = G + (size_t)t * 64 * 256;
#pragma unroll
  for (int j = 0; j < 8; j++) {
    const int col = (nh * 8 + j) * 16 + ln;
    const float bb = bias[col];
#pragma unroll
    for (int v = 0; v < 8; v++) {
      const int br = 16 * m + 8 * hh + v;
      gp[br * 256 + col] = acc[j].f[v] + bb;
    }
  }
}

// ---------------- layer-1 input gates: G = Y1 @ Wih1^T + b1 ----------------
__global__ __launch_bounds__(256) void gates1_kernel(
    const _Float16* __restrict__ Y1, const _Float16* __restrict__ Wih,
    const float* __restrict__ bias, float* __restrict__ G)
{
  const int t = blockIdx.x;
  const int tid = threadIdx.x, lane = tid & 31, w = tid >> 5;
  const int m = w >> 1, nh = w & 1;
  const int ln = lane & 15, hh = lane >> 4;
  const int b = 16 * m + ln;

  const _Float16* ar = Y1 + ((size_t)t * 64 + b) * 64;

  F8 acc[8];
#pragma unroll
  for (int j = 0; j < 8; j++)
#pragma unroll
    for (int v = 0; v < 8; v++) acc[j].f[v] = 0.0f;

#pragma unroll
  for (int s = 0; s < 2; s++) {           // K = 64 = 2 x 32
    const int k0 = 32 * s;
    H16 a;
    a.h8[0] = *(const v8h*)(ar + k0 + hh * 16);
    a.h8[1] = *(const v8h*)(ar + k0 + hh * 16 + 8);
#pragma unroll
    for (int j = 0; j < 8; j++) {
      const int col0 = (nh * 8 + j) * 16;
      v16h bf = *(const v16h*)(Wih + (size_t)(col0 + ln) * 64 + k0 + hh * 16);
      acc[j].v = wmma_f16(a.v, bf, acc[j].v);
    }
  }

  float* gp = G + (size_t)t * 64 * 256;
#pragma unroll
  for (int j = 0; j < 8; j++) {
    const int col = (nh * 8 + j) * 16 + ln;
    const float bb = bias[col];
#pragma unroll
    for (int v = 0; v < 8; v++) {
      const int br = 16 * m + 8 * hh + v;
      gp[br * 256 + col] = acc[j].f[v] + bb;
    }
  }
}

// ---------------- persistent LSTM recurrence ----------------
// 4 blocks x 128 threads; block handles batch rows [16*blk, 16*blk+16).
// Wave w owns h-tile w (h in [16w,16w+16)) and all 4 gate tiles -> i,f,g,o
// for each (b,h) live in one wave; cell state c stays in registers.
// Gate slices G[t] are double-buffered in LDS via async global->LDS copies:
// step t+1's 16KB slice streams in while step t computes; consumption is
// gated on s_wait_asynccnt 8 (in-order completion).
__global__ __launch_bounds__(128) void lstm_layer_kernel(
    const float* __restrict__ G, const _Float16* __restrict__ Whh,
    _Float16* __restrict__ Y, float* __restrict__ hlast, int T)
{
  __shared__ __align__(16) _Float16 sH[16 * 64];   // h_{t-1} for 16 batch rows
  __shared__ __align__(16) float    sG[2][16 * 256]; // double-buffered gates

  const int tid = threadIdx.x, lane = tid & 31, w = tid >> 5;   // w = h-tile 0..3
  const int ln = lane & 15, hh = lane >> 4;
  const int bblk = blockIdx.x * 16;
  const int hcol = 16 * w + ln;

  for (int i = tid; i < 16 * 64; i += 128) sH[i] = (_Float16)0.0f;

  float c[8];
#pragma unroll
  for (int v = 0; v < 8; v++) c[v] = 0.0f;

  // Whh fragments resident in registers for the whole recurrence:
  // B column = 64*gate + hcol, K split 2 x 32.
  v16h bf[4][2];
#pragma unroll
  for (int g = 0; g < 4; g++)
#pragma unroll
    for (int s = 0; s < 2; s++)
      bf[g][s] = *(const v16h*)(Whh + (size_t)(64 * g + hcol) * 64 + 32 * s + hh * 16);

  // async-copy addressing: lane owns 8 x 16B chunks, stride 2KB (immediates)
  const unsigned gvofs  = (unsigned)tid * 16u;   // byte offset into 16KB slice
  const unsigned ldsG0 = (unsigned)(uintptr_t)(&sG[0][0]) + (unsigned)tid * 16u;
  const unsigned ldsG1 = (unsigned)(uintptr_t)(&sG[1][0]) + (unsigned)tid * 16u;

  // prologue: stage t=0 gates into buffer 0
  stage_gates(ldsG0, gvofs, G + (size_t)bblk * 256);
  __syncthreads();

  for (int t = 0; t < T; t++) {
    // A = h_{t-1} [16 x 64] from LDS
    H16 a0, a1;
    a0.h8[0] = *(const v8h*)(sH + ln * 64 + 0  + hh * 16);
    a0.h8[1] = *(const v8h*)(sH + ln * 64 + 8  + hh * 16);
    a1.h8[0] = *(const v8h*)(sH + ln * 64 + 32 + hh * 16);
    a1.h8[1] = *(const v8h*)(sH + ln * 64 + 40 + hh * 16);

    F8 acc[4];
#pragma unroll
    for (int g = 0; g < 4; g++) {
      F8 z;
#pragma unroll
      for (int v = 0; v < 8; v++) z.f[v] = 0.0f;
      z.v = wmma_f16(a0.v, bf[g][0], z.v);
      z.v = wmma_f16(a1.v, bf[g][1], z.v);
      acc[g] = z;
    }

    // stream step t+1's gate slice while this step finishes
    if (t + 1 < T) {
      stage_gates(((t + 1) & 1) ? ldsG1 : ldsG0, gvofs,
                  G + (size_t)(t + 1) * 64 * 256 + (size_t)bblk * 256);
      // 16 outstanding; <=8 left  ==>  step t's 8 transfers have landed
      asm volatile("s_wait_asynccnt 0x8" ::: "memory");
    } else {
      asm volatile("s_wait_asynccnt 0x0" ::: "memory");
    }
    __syncthreads();                       // sH reads done + gate slice visible

    const float* gb = &sG[t & 1][0];
#pragma unroll
    for (int v = 0; v < 8; v++) {
      const int br = 8 * hh + v;           // local batch row
      const float* gr = gb + br * 256;
      float i_ = acc[0].f[v] + gr[hcol];
      float f_ = acc[1].f[v] + gr[64 + hcol];
      float gg = acc[2].f[v] + gr[128 + hcol];
      float o_ = acc[3].f[v] + gr[192 + hcol];
      float cn = sigm(f_) * c[v] + sigm(i_) * tanhf(gg);
      c[v] = cn;
      float hv = sigm(o_) * tanhf(cn);
      _Float16 hf = (_Float16)hv;
      sH[br * 64 + hcol] = hf;
      Y[((size_t)t * 64 + (bblk + br)) * 64 + hcol] = hf;
      if (hlast && t == T - 1) hlast[(bblk + br) * 64 + hcol] = hv;
    }
    __syncthreads();                       // h_t visible before step t+1 reads
  }
}

// ---------------- FC head: relu(h @ W1^T + b1) @ W2^T + b2 ----------------
__global__ __launch_bounds__(256) void head_kernel(
    const float* __restrict__ hl, const float* __restrict__ W1, const float* __restrict__ bfc1,
    const float* __restrict__ W2, const float* __restrict__ bfc2, float* __restrict__ out)
{
  __shared__ float sHl[64 * 64];
  __shared__ float sZ[64 * 64];
  const int tid = threadIdx.x;
  for (int i = tid; i < 4096; i += 256) sHl[i] = hl[i];
  __syncthreads();
  for (int i = tid; i < 4096; i += 256) {
    const int b = i >> 6, j = i & 63;
    float s = bfc1[j];
    for (int k = 0; k < 64; k++) s += sHl[b * 64 + k] * W1[j * 64 + k];
    sZ[b * 64 + j] = fmaxf(s, 0.0f);
  }
  __syncthreads();
  for (int i = tid; i < 512; i += 256) {
    const int b = i >> 3, o = i & 7;
    float s = bfc2[o];
    for (int j = 0; j < 64; j++) s += sZ[b * 64 + j] * W2[o * 64 + j];
    out[b * 8 + o] = s;
  }
}

extern "C" void kernel_launch(void* const* d_in, const int* in_sizes, int n_in,
                              void* d_out, int out_size, void* d_ws, size_t ws_size,
                              hipStream_t stream) {
  const float* XA   = (const float*)d_in[0];
  const float* XB   = (const float*)d_in[1];
  const float* Wih0 = (const float*)d_in[2];
  const float* Whh0 = (const float*)d_in[3];
  const float* b0   = (const float*)d_in[4];
  const float* Wih1 = (const float*)d_in[5];
  const float* Whh1 = (const float*)d_in[6];
  const float* b1l  = (const float*)d_in[7];
  const float* W1   = (const float*)d_in[8];
  const float* bfc1 = (const float*)d_in[9];
  const float* W2   = (const float*)d_in[10];
  const float* bfc2 = (const float*)d_in[11];
  float* out = (float*)d_out;

  // workspace layout (all offsets 256B-aligned)
  char* ws = (char*)d_ws;
  _Float16* Wih0h = (_Float16*)(ws);                        //  64 KB
  _Float16* Whh0h = (_Float16*)(ws + 65536);                //  32 KB
  _Float16* Wih1h = (_Float16*)(ws + 98304);                //  32 KB
  _Float16* Whh1h = (_Float16*)(ws + 131072);               //  32 KB
  float*    hlast = (float*)   (ws + 163840);               //  16 KB
  _Float16* Y1    = (_Float16*)(ws + 180224);               //  32 MB
  _Float16* Y2    = (_Float16*)(ws + 180224 + 33554432);    //  32 MB
  float*    G     = (float*)   (ws + 180224 + 2ull * 33554432); // 256 MB (reused by both layers)

  cvt_f16_kernel<<<128, 256, 0, stream>>>(Wih0, Wih0h, 32768);
  cvt_f16_kernel<<<64,  256, 0, stream>>>(Whh0, Whh0h, 16384);
  cvt_f16_kernel<<<64,  256, 0, stream>>>(Wih1, Wih1h, 16384);
  cvt_f16_kernel<<<64,  256, 0, stream>>>(Whh1, Whh1h, 16384);

  gates0_kernel<<<4096, 256, 0, stream>>>(XA, XB, Wih0h, b0, G);
  lstm_layer_kernel<<<4, 128, 0, stream>>>(G, Whh0h, Y1, nullptr, 4096);
  gates1_kernel<<<4096, 256, 0, stream>>>(Y1, Wih1h, b1l, G);
  lstm_layer_kernel<<<4, 128, 0, stream>>>(G, Whh1h, Y2, hlast, 4096);
  head_kernel<<<1, 256, 0, stream>>>(hlast, W1, bfc1, W2, bfc2, out);

  (void)in_sizes; (void)n_in; (void)out_size; (void)ws_size;
}
